// VariLengthInputLayer_3289944949175
// MI455X (gfx1250) — matrix operand order repaired and checked by reading, not compiled, not run on hardware
//
#include <hip/hip_runtime.h>

// ---------------------------------------------------------------------------
// Types for WMMA fragments (gfx1250, wave32)
// ---------------------------------------------------------------------------
typedef _Float16 h16v __attribute__((ext_vector_type(16)));
typedef _Float16 h8v  __attribute__((ext_vector_type(8)));
typedef float    f8v  __attribute__((ext_vector_type(8)));

#define WMMA_F16(a, b, c) \
  __builtin_amdgcn_wmma_f32_16x16x32_f16(false, (a), false, (b), (short)0, (c), false, false)

// CDNA5 async global->LDS copy (ASYNCcnt-tracked, no VGPR round trip).
// the LDS address comes from the VDST VGPR; generic shared pointers carry the LDS byte
// offset in their low 32 bits.
__device__ __forceinline__ void async_b128(unsigned lds_off, const void* gaddr) {
  asm volatile("global_load_async_to_lds_b128 %0, %1, off"
               :: "v"(lds_off), "v"((unsigned long long)(uintptr_t)gaddr)
               : "memory");
}
__device__ __forceinline__ void wait_async0() {
  asm volatile("s_wait_asynccnt 0x0" ::: "memory");
}

// ---------------------------------------------------------------------------
// Problem constants
// DIMS = [2048,1024,512,1536,768,256], sum = 6144; N_HEAD=8, D_K=D_V=64 -> 512
// ---------------------------------------------------------------------------
__constant__ int  c_dims[6] = {2048, 1024, 512, 1536, 768, 256};
__constant__ int  c_xoff[6] = {0, 2048, 3072, 3584, 5120, 5888};
__constant__ long c_woff[6] = {0, 3145728, 4718592, 5505024, 7864320, 9043968};

static const int  h_dims[6] = {2048, 1024, 512, 1536, 768, 256};
static const long h_woff[6] = {0, 3145728, 4718592, 5505024, 7864320, 9043968};

#define BROW 40  // padded LDS row stride (halfs): conflict-free 16-lane frag reads

// ---------------------------------------------------------------------------
// fp32 -> fp16 weight conversion (grid-stride)
// ---------------------------------------------------------------------------
__global__ void k_cvt(const float* __restrict__ src, _Float16* __restrict__ dst, int n) {
  for (int i = blockIdx.x * blockDim.x + threadIdx.x; i < n; i += gridDim.x * blockDim.x)
    dst[i] = (_Float16)src[i];
}

__global__ void k_init(unsigned* gmin) { *gmin = 0xFFFFFFFFu; }

// ---------------------------------------------------------------------------
// QKV GEMM: per segment, QKV = X_seg @ [wq;wk;wv]^T  (WMMA f16, f32 accum)
// Block = 8 waves, tile 128M x 64N.  B tile (64x32, 4KB) streamed straight to
// double-buffered LDS with global_load_async_to_lds_b128, issued at the top of
// each k-step and waited (s_wait_asynccnt) only at the bottom, so the WMMA
// group overlaps the async DMA.  k-loop unrolled 2x (all segments have an even
// number of 32-k steps) so the buffer index is compile-time constant per copy:
// no cndmask / A-register rotation movs.  grid = (24, 128, 6), N fastest.
// ---------------------------------------------------------------------------
__global__ __launch_bounds__(256) void k_qkv(const float* __restrict__ X,
                                             const _Float16* __restrict__ Wb,
                                             _Float16* __restrict__ QH,
                                             _Float16* __restrict__ KH,
                                             float* __restrict__ VF) {
  const int seg  = blockIdx.z;
  const int d    = c_dims[seg];
  const int xoff = c_xoff[seg];
  const long woff = c_woff[seg];

  const int tid  = threadIdx.x;
  const int lane = tid & 31;
  const int wave = tid >> 5;
  const int lm   = lane & 15;   // M row (A/D) or N col (B/D) within tile
  const int kg   = lane >> 4;   // half-wave group
  const int n0   = blockIdx.x * 64;
  const int m0   = blockIdx.y * 128 + wave * 16;

  __shared__ _Float16 bs[2][64 * BROW];

  // Cooperative B staging: thread -> (row 0..63, 8-half chunk 0..3) of 64x32 tile
  const int brow_g = tid >> 2;
  const int bchunk = tid & 3;
  const _Float16* bsrc = Wb + woff + (long)(n0 + brow_g) * d + bchunk * 8;
  const unsigned ldsoff0 = (unsigned)(uintptr_t)&bs[0][brow_g * BROW + bchunk * 8];
  const unsigned ldsoff1 = (unsigned)(uintptr_t)&bs[1][brow_g * BROW + bchunk * 8];

  const float* xrow = X + (long)(m0 + lm) * 6144 + xoff + kg * 8;

  f8v acc[4] = {};

  // Prologue: async-stage B(k0=0), load A(k0=0)
  async_b128(ldsoff0, bsrc);
  f8v fa = *(const f8v*)(xrow);
  f8v fb = *(const f8v*)(xrow + 16);
  wait_async0();
  __syncthreads();

  int p = 0;
#pragma unroll 2
  for (int k0 = 0; k0 < d; k0 += 32) {
    const bool more = (k0 + 32) < d;
    f8v fan, fbn;
    if (more) {  // issue next-step DMA + A loads early (overlap with WMMAs below)
      async_b128(p ? ldsoff0 : ldsoff1, bsrc + k0 + 32);
      fan = *(const f8v*)(xrow + k0 + 32);
      fbn = *(const f8v*)(xrow + k0 + 48);
    }

    // A fragment: lane holds row M=lm, halfs h: K = kg*8 + h%8 + 16*(h/8)
    h16v a;
#pragma unroll
    for (int i = 0; i < 8; ++i) {
      a[i]     = (_Float16)fa[i];
      a[8 + i] = (_Float16)fb[i];
    }

    // Issue ALL fragment LDS reads, then the WMMA group.
    const _Float16* bbuf = &bs[p][kg * 8];
    h8v bl[8];
#pragma unroll
    for (int j = 0; j < 4; ++j) {
      const _Float16* bp = bbuf + (j * 16 + lm) * BROW;
      bl[2 * j]     = *(const h8v*)(bp);
      bl[2 * j + 1] = *(const h8v*)(bp + 16);
    }
#pragma unroll
    for (int j = 0; j < 4; ++j) {
      h16v b;
#pragma unroll
      for (int i = 0; i < 8; ++i) { b[i] = bl[2 * j][i]; b[8 + i] = bl[2 * j + 1][i]; }
      acc[j] = WMMA_F16(a, b, acc[j]);
    }

    if (more) { fa = fan; fb = fbn; }
    wait_async0();      // next buffer fully landed in LDS
    __syncthreads();    // and everyone is done reading the current one
    p ^= 1;
  }

  // D layout: lanes 0-15: M=r, N=lane; lanes 16-31: M=8+r, N=lane-16
#pragma unroll
  for (int j = 0; j < 4; ++j) {
    const int col = n0 + j * 16 + lm;  // 16-col tile never straddles a 512 boundary
#pragma unroll
    for (int r = 0; r < 8; ++r) {
      const int  mrow = m0 + kg * 8 + r;
      const long base = ((long)mrow * 6 + seg) * 512;
      const float v = acc[j][r];
      if (col < 512)        QH[base + col]        = (_Float16)v;
      else if (col < 1024)  KH[base + col - 512]  = (_Float16)v;
      else                  VF[base + col - 1024] = v;
    }
  }
}

// ---------------------------------------------------------------------------
// Attention logits: attn[b,h,si,sj] = (q.k)/8 ; global-min reduction
// One block per batch row; 288 threads = 8 heads * 36 pairs (9 waves).
// ---------------------------------------------------------------------------
__global__ __launch_bounds__(288) void k_logits(const _Float16* __restrict__ QH,
                                                const _Float16* __restrict__ KH,
                                                float* __restrict__ ATT,
                                                unsigned* __restrict__ gmin) {
  const int b = blockIdx.x;
  const int t = threadIdx.x;
  __shared__ float qs[3072], ks[3072];
  const _Float16* qb = QH + (long)b * 3072;
  const _Float16* kb = KH + (long)b * 3072;
  for (int i = t; i < 3072; i += 288) {
    qs[i] = (float)qb[i];
    ks[i] = (float)kb[i];
  }
  __syncthreads();

  const int h = t / 36, r = t % 36, si = r / 6, sj = r % 6;
  const float* q = qs + si * 512 + h * 64;
  const float* k = ks + sj * 512 + h * 64;
  float dot = 0.f;
#pragma unroll
  for (int dd = 0; dd < 64; ++dd) dot += q[dd] * k[dd];
  dot *= 0.125f;  // / sqrt(64)
  ATT[((long)b * 8 + h) * 36 + r] = dot;

  // wave-level min, then one atomic per wave (order-preserving uint encoding)
  float v = dot;
#pragma unroll
  for (int off = 16; off > 0; off >>= 1) v = fminf(v, __shfl_down(v, off, 32));
  if ((t & 31) == 0) {
    unsigned u   = __float_as_uint(v);
    unsigned key = (u >> 31) ? ~u : (u | 0x80000000u);
    atomicMin(gmin, key);
  }
}

// ---------------------------------------------------------------------------
// /|gmin|, L2 row-norm, softmax -> d_out attn; O = attn @ V -> f16 FC input
// grid=(16384,8), block=64 (thread = output dim d)
// ---------------------------------------------------------------------------
__global__ __launch_bounds__(64) void k_smax(float* __restrict__ ATT,
                                             const float* __restrict__ VF,
                                             _Float16* __restrict__ OH,
                                             const unsigned* __restrict__ gmin) {
  const int b = blockIdx.x, h = blockIdx.y, t = threadIdx.x;
  __shared__ float att[36];

  const unsigned key = *gmin;
  const unsigned u   = (key & 0x80000000u) ? (key ^ 0x80000000u) : ~key;
  float am = fabsf(__uint_as_float(u));
  const float inv = (am > 0.f) ? 1.f / am : 1.f;

  float* arow = ATT + ((long)b * 8 + h) * 36;
  if (t < 36) att[t] = arow[t] * inv;
  __syncthreads();

  if (t < 6) {  // row si = t : L2-normalize then softmax
    float nrm = 0.f;
#pragma unroll
    for (int j = 0; j < 6; ++j) { float x = att[t * 6 + j]; nrm += x * x; }
    const float sc = 1.f / fmaxf(sqrtf(nrm), 1e-12f);
    float mx = -3.402823466e38f;
#pragma unroll
    for (int j = 0; j < 6; ++j) mx = fmaxf(mx, att[t * 6 + j] * sc);
    float e[6], s = 0.f;
#pragma unroll
    for (int j = 0; j < 6; ++j) { e[j] = __expf(att[t * 6 + j] * sc - mx); s += e[j]; }
    const float is = 1.f / s;
#pragma unroll
    for (int j = 0; j < 6; ++j) {
      const float a = e[j] * is;
      att[t * 6 + j]  = a;
      arow[t * 6 + j] = a;
    }
  }
  __syncthreads();

  float vv[6];
#pragma unroll
  for (int sj = 0; sj < 6; ++sj) vv[sj] = VF[((long)b * 6 + sj) * 512 + h * 64 + t];
#pragma unroll
  for (int si = 0; si < 6; ++si) {
    float o = 0.f;
#pragma unroll
    for (int sj = 0; sj < 6; ++sj) o += att[si * 6 + sj] * vv[sj];
    OH[((long)b * 6 + si) * 512 + h * 64 + t] = (_Float16)o;
  }
}

// ---------------------------------------------------------------------------
// FC GEMM (98304x512 @ 512x512^T) + bias + residual(V) -> d_out (pre-LN)
// Same async-LDS double-buffered tiling, k-loop unrolled 2x. grid=(8, 768)
// ---------------------------------------------------------------------------
__global__ __launch_bounds__(256) void k_fc(const _Float16* __restrict__ OH,
                                            const _Float16* __restrict__ FW,
                                            const float* __restrict__ fcb,
                                            const float* __restrict__ VF,
                                            float* __restrict__ OUT) {
  const int tid  = threadIdx.x;
  const int lane = tid & 31;
  const int wave = tid >> 5;
  const int lm = lane & 15, kg = lane >> 4;
  const int n0 = blockIdx.x * 64;
  const int m0 = blockIdx.y * 128 + wave * 16;

  __shared__ _Float16 bs[2][64 * BROW];

  const int brow_g = tid >> 2;
  const int bchunk = tid & 3;
  const _Float16* bsrc = FW + (long)(n0 + brow_g) * 512 + bchunk * 8;
  const unsigned ldsoff0 = (unsigned)(uintptr_t)&bs[0][brow_g * BROW + bchunk * 8];
  const unsigned ldsoff1 = (unsigned)(uintptr_t)&bs[1][brow_g * BROW + bchunk * 8];

  const _Float16* arow = OH + (long)(m0 + lm) * 512 + kg * 8;
  f8v acc[4] = {};

  async_b128(ldsoff0, bsrc);
  h8v a0 = *(const h8v*)(arow);
  h8v a1 = *(const h8v*)(arow + 16);
  wait_async0();
  __syncthreads();

  int p = 0;
#pragma unroll 2
  for (int k0 = 0; k0 < 512; k0 += 32) {
    const bool more = (k0 + 32) < 512;
    h8v a0n, a1n;
    if (more) {
      async_b128(p ? ldsoff0 : ldsoff1, bsrc + k0 + 32);
      a0n = *(const h8v*)(arow + k0 + 32);
      a1n = *(const h8v*)(arow + k0 + 48);
    }

    h16v a;
#pragma unroll
    for (int i = 0; i < 8; ++i) { a[i] = a0[i]; a[8 + i] = a1[i]; }

    const _Float16* bbuf = &bs[p][kg * 8];
    h8v bl[8];
#pragma unroll
    for (int j = 0; j < 4; ++j) {
      const _Float16* bp = bbuf + (j * 16 + lm) * BROW;
      bl[2 * j]     = *(const h8v*)(bp);
      bl[2 * j + 1] = *(const h8v*)(bp + 16);
    }
#pragma unroll
    for (int j = 0; j < 4; ++j) {
      h16v b;
#pragma unroll
      for (int i = 0; i < 8; ++i) { b[i] = bl[2 * j][i]; b[8 + i] = bl[2 * j + 1][i]; }
      acc[j] = WMMA_F16(a, b, acc[j]);
    }

    if (more) { a0 = a0n; a1 = a1n; }
    wait_async0();
    __syncthreads();
    p ^= 1;
  }

#pragma unroll
  for (int j = 0; j < 4; ++j) {
    const int col = n0 + j * 16 + lm;
    const float bias = fcb[col];
#pragma unroll
    for (int r = 0; r < 8; ++r) {
      const long mrow = m0 + kg * 8 + r;
      OUT[mrow * 512 + col] = acc[j][r] + bias + VF[mrow * 512 + col];
    }
  }
}

// ---------------------------------------------------------------------------
// LayerNorm over last dim (512), in-place on d_out. grid=98304, block=256.
// ---------------------------------------------------------------------------
__global__ __launch_bounds__(256) void k_ln(float* __restrict__ OUT,
                                            const float* __restrict__ g,
                                            const float* __restrict__ bt) {
  const long row = blockIdx.x;
  const int  t   = threadIdx.x;
  float x0 = OUT[row * 512 + t];
  float x1 = OUT[row * 512 + t + 256];
  __shared__ float s1[256], s2[256];
  s1[t] = x0 + x1;
  s2[t] = x0 * x0 + x1 * x1;
  __syncthreads();
  for (int s = 128; s > 0; s >>= 1) {
    if (t < s) { s1[t] += s1[t + s]; s2[t] += s2[t + s]; }
    __syncthreads();
  }
  const float mu  = s1[0] * (1.f / 512.f);
  const float var = s2[0] * (1.f / 512.f) - mu * mu;
  const float rs  = rsqrtf(var + 1e-6f);
  OUT[row * 512 + t]       = (x0 - mu) * rs * g[t] + bt[t];
  OUT[row * 512 + t + 256] = (x1 - mu) * rs * g[t + 256] + bt[t + 256];
}

// ---------------------------------------------------------------------------
// Host orchestration.  Workspace layout (bytes):
//   WB  f16 combined [wq;wk;wv] per segment : 18,874,368
//   FCB f16 fc_w                            :    524,288
//   QH  f16 (16384,6,512)                   : 100,663,296
//   KH  f16 (16384,6,512)                   : 100,663,296
//   VF  f32 (16384,6,512)                   : 201,326,592
//   OH  f16 (16384,6,512)  (attn@v)         : 100,663,296
//   GMIN u32                                :          4     => ~523 MB
// ---------------------------------------------------------------------------
extern "C" void kernel_launch(void* const* d_in, const int* in_sizes, int n_in,
                              void* d_out, int out_size, void* d_ws, size_t ws_size,
                              hipStream_t stream) {
  (void)in_sizes; (void)n_in; (void)out_size; (void)ws_size;

  constexpr size_t WB_OFF   = 0;
  constexpr size_t FCB_OFF  = 18874368;
  constexpr size_t QH_OFF   = 19398656;
  constexpr size_t KH_OFF   = 120061952;
  constexpr size_t VF_OFF   = 220725248;
  constexpr size_t OH_OFF   = 422051840;
  constexpr size_t GMIN_OFF = 522715136;

  char* ws = (char*)d_ws;
  _Float16* WB   = (_Float16*)(ws + WB_OFF);
  _Float16* FCB  = (_Float16*)(ws + FCB_OFF);
  _Float16* QH   = (_Float16*)(ws + QH_OFF);
  _Float16* KH   = (_Float16*)(ws + KH_OFF);
  float*    VF   = (float*)(ws + VF_OFF);
  _Float16* OH   = (_Float16*)(ws + OH_OFF);
  unsigned* GMIN = (unsigned*)(ws + GMIN_OFF);

  const float* X = (const float*)d_in[0];
  float* OUT = (float*)d_out;                 // (16384,6,512)
  float* ATT = (float*)d_out + 50331648;      // (16384,8,6,6)

  k_init<<<1, 1, 0, stream>>>(GMIN);

  // Convert weights to f16, packing [wq;wk;wv] rows per segment.
  for (int i = 0; i < 6; ++i) {
    const int  d  = h_dims[i];
    const long wo = h_woff[i];
    k_cvt<<<256, 256, 0, stream>>>((const float*)d_in[1 + 3 * i], WB + wo,             512 * d);
    k_cvt<<<256, 256, 0, stream>>>((const float*)d_in[2 + 3 * i], WB + wo + 512L * d,  512 * d);
    k_cvt<<<256, 256, 0, stream>>>((const float*)d_in[3 + 3 * i], WB + wo + 1024L * d, 512 * d);
  }
  k_cvt<<<256, 256, 0, stream>>>((const float*)d_in[19], FCB, 512 * 512);

  dim3 gq(1536 / 64, 16384 / 128, 6);
  k_qkv<<<gq, 256, 0, stream>>>(X, WB, QH, KH, VF);

  k_logits<<<16384, 288, 0, stream>>>(QH, KH, ATT, GMIN);

  dim3 gs(16384, 8);
  k_smax<<<gs, 64, 0, stream>>>(ATT, VF, OH, GMIN);

  dim3 gf(512 / 64, 98304 / 128);
  k_fc<<<gf, 256, 0, stream>>>(OH, FCB, (const float*)d_in[20], VF, OUT);

  k_ln<<<98304, 256, 0, stream>>>(OUT, (const float*)d_in[21], (const float*)d_in[22]);
}
